// DyadXLSTM_24326694764772
// MI455X (gfx1250) — matrix operand order repaired and checked by reading, not compiled, not run on hardware
//
#include <hip/hip_runtime.h>

// xLSTM forward (DyadXLSTM) for gfx1250 / MI455X.
#define BB   32
#define TT   256
#define FF   224
#define EE   32
#define DD   256   // block width (F+E)
#define NHH  4
#define DHH  64
#define HH   256   // NH*DH

typedef __attribute__((ext_vector_type(16))) __bf16 v16bf;
typedef __attribute__((ext_vector_type(8)))  float  v8f;

__device__ __forceinline__ unsigned short f32_to_bf16(float f) {
  union { float f; unsigned int u; } c; c.f = f;
  unsigned int u = c.u;
  unsigned int r = u + 0x7FFFu + ((u >> 16) & 1u);   // round-to-nearest-even
  if ((u & 0x7F800000u) == 0x7F800000u) r = u;       // preserve inf/nan
  return (unsigned short)(r >> 16);
}

// ---------------------------------------------------------------------------
// x = concat(X, embed[dyad_idx] broadcast over T)
// ---------------------------------------------------------------------------
__global__ void concat_kernel(const float* __restrict__ X, const int* __restrict__ dyad,
                              const float* __restrict__ embed, float* __restrict__ x) {
  int idx = blockIdx.x * blockDim.x + threadIdx.x;
  if (idx >= BB * TT * DD) return;
  int c  = idx & (DD - 1);
  int bt = idx >> 8;          // / DD
  int b  = bt / TT;
  int t  = bt % TT;
  x[idx] = (c < FF) ? X[((size_t)b * TT + t) * FF + c]
                    : embed[(size_t)dyad[b] * EE + (c - FF)];
}

// ---------------------------------------------------------------------------
// Row LayerNorm over D=256 (one block per row), writes f32 + bf16 copies.
// ---------------------------------------------------------------------------
__global__ void ln_kernel(const float* __restrict__ x, const float* __restrict__ g,
                          const float* __restrict__ bta, float* __restrict__ xn,
                          unsigned short* __restrict__ xnb) {
  int row = blockIdx.x;
  int t = threadIdx.x;
  float v = x[(size_t)row * DD + t];
  __shared__ float red[DD];
  red[t] = v;
  __syncthreads();
  for (int s = DD / 2; s > 0; s >>= 1) {
    if (t < s) red[t] += red[t + s];
    __syncthreads();
  }
  float mu = red[0] * (1.0f / DD);
  __syncthreads();
  float dv = v - mu;
  red[t] = dv * dv;
  __syncthreads();
  for (int s = DD / 2; s > 0; s >>= 1) {
    if (t < s) red[t] += red[t + s];
    __syncthreads();
  }
  float var = red[0] * (1.0f / DD);
  float y = dv * rsqrtf(var + 1e-5f) * g[t] + bta[t];
  xn[(size_t)row * DD + t] = y;
  xnb[(size_t)row * DD + t] = f32_to_bf16(y);
}

// ---------------------------------------------------------------------------
// Generic f32 -> bf16(bits) conversion (activations: h buffer)
// ---------------------------------------------------------------------------
__global__ void f2bf_kernel(const float* __restrict__ src, unsigned short* __restrict__ dst, int n) {
  int i = blockIdx.x * blockDim.x + threadIdx.x;
  if (i < n) dst[i] = f32_to_bf16(src[i]);
}

// ---------------------------------------------------------------------------
// Weight pre-pack: f32 [256,256] row-major -> bf16 WMMA-B-fragment-linear
// dwords. dword idx = ((ntile*8 + kblock)*32 + lane)*8 + j, each dword packs
// rows (kblock*32 + hi*16 + 2j, +1) at column ntile*16 + (lane&15).
// GEMM then reads each fragment as 8 consecutive dwords per lane (-> b128s).
// ---------------------------------------------------------------------------
__global__ void pack_w_kernel(const float* __restrict__ W, unsigned int* __restrict__ Wp) {
  int idx = blockIdx.x * blockDim.x + threadIdx.x;   // 16*8*32*8 = 32768
  if (idx >= 32768) return;
  int j    = idx & 7;
  int lane = (idx >> 3) & 31;
  int kb   = (idx >> 8) & 7;
  int nt   = idx >> 11;
  int ncol = nt * 16 + (lane & 15);
  int hi   = lane >> 4;
  int r0   = kb * 32 + hi * 16 + (j << 1);
  unsigned int lo = f32_to_bf16(W[(size_t)r0 * HH + ncol]);
  unsigned int h2 = f32_to_bf16(W[(size_t)(r0 + 1) * HH + ncol]);
  Wp[idx] = lo | (h2 << 16);
}

// ---------------------------------------------------------------------------
// WMMA GEMM: [8192,256] (bf16) x [256,256] (packed bf16) -> f32.
// One wave computes a 16(M) x 64(N) strip: A fragment reused across 4 WMMAs
// per K-step, K fully unrolled (8 steps -> 32 v_wmma per kernel).
// A/B k order contiguous (k = 16*hi + s, same in both), C/D per CDNA5 ISA 7.12.2 (16-bit A 16x32, B 32x16, C/D f32).
// MODE: 0=plain, 1=*1/8 (k-scale), 2=sigmoid (o-gate), 3=+res (residual Wp)
// ---------------------------------------------------------------------------
template <int MODE>
__global__ void gemm256_wmma(const unsigned short* __restrict__ A,
                             const unsigned int* __restrict__ Bp,
                             float* __restrict__ C,
                             const float* __restrict__ res) {
  const int lane = threadIdx.x & 31;
  const int wave = threadIdx.x >> 5;        // 0..3 -> N strip of 64
  const int tm = blockIdx.x;                // 0..511 (M/16)
  const int l15 = lane & 15;
  const int hi = lane >> 4;
  const int mrow = tm * 16 + l15;

  union AF { v16bf v; unsigned int u[8]; };
  union CF { v8f v; float f[8]; };
  CF acc[4];
#pragma unroll
  for (int s = 0; s < 4; ++s)
#pragma unroll
    for (int i = 0; i < 8; ++i) acc[s].f[i] = 0.f;

  const unsigned int* Arow = (const unsigned int*)(A + (size_t)mrow * DD);

#pragma unroll
  for (int kb = 0; kb < 8; ++kb) {
    AF a;
#pragma unroll
    for (int j = 0; j < 4; ++j) a.u[j]     = Arow[kb * 16 + (hi << 3) + j];
#pragma unroll
    for (int j = 0; j < 4; ++j) a.u[4 + j] = Arow[kb * 16 + (hi << 3) + 4 + j];
#pragma unroll
    for (int s = 0; s < 4; ++s) {
      const int nt = wave * 4 + s;
      const unsigned int* bp = Bp + ((((size_t)nt * 8 + kb) * 32 + lane) << 3);
      AF b;
#pragma unroll
      for (int j = 0; j < 8; ++j) b.u[j] = bp[j];
      acc[s].v = __builtin_amdgcn_wmma_f32_16x16x32_bf16(false, a.v, false, b.v,
                                                         (short)0, acc[s].v, false, false);
    }
  }

#pragma unroll
  for (int s = 0; s < 4; ++s) {
    const int ncol = wave * 64 + s * 16 + l15;
#pragma unroll
    for (int r = 0; r < 8; ++r) {
      int row = tm * 16 + (hi << 3) + r;
      size_t oidx = (size_t)row * HH + ncol;
      float vv = acc[s].f[r];
      if (MODE == 1)      vv *= 0.125f;                    // DH^-0.5, DH=64
      else if (MODE == 2) vv = 1.f / (1.f + __expf(-vv));  // sigmoid
      else if (MODE == 3) vv += res[oidx];                 // residual
      C[oidx] = vv;
    }
  }
}

// ---------------------------------------------------------------------------
// ip/fp scalar gate preactivations: xn @ Wi + bi, xn @ Wf + bf  -> [B,T,NH]
// ---------------------------------------------------------------------------
__global__ void ipfp_kernel(const float* __restrict__ xn, const float* __restrict__ Wi,
                            const float* __restrict__ Wf, const float* __restrict__ bi,
                            const float* __restrict__ bf_, float* __restrict__ ip,
                            float* __restrict__ fp) {
  int idx = blockIdx.x * blockDim.x + threadIdx.x;
  if (idx >= BB * TT * NHH) return;
  int hd = idx & (NHH - 1);
  int row = idx >> 2;
  const float* xr = xn + (size_t)row * DD;
  float ai = 0.f, af = 0.f;
  for (int kk = 0; kk < DD; ++kk) {
    float xv = xr[kk];
    ai += xv * Wi[kk * NHH + hd];
    af += xv * Wf[kk * NHH + hd];
  }
  ip[idx] = ai + bi[hd];
  fp[idx] = af + bf_[hd];
}

// ---------------------------------------------------------------------------
// mLSTM sequential scan. One block per (b,head); 256 threads.
// 64x64 matrix memory C in registers: thread (d = t&63, cb = t>>6) owns
// C[d][cb*16 .. cb*16+15]. Per step: stabilized exp gating, rank-1 update,
// matvec num = C q with cross-thread LDS reduction.
// ---------------------------------------------------------------------------
__global__ void mlstm_scan(const float* __restrict__ q, const float* __restrict__ k,
                           const float* __restrict__ v, const float* __restrict__ o,
                           const float* __restrict__ ip, const float* __restrict__ fp,
                           float* __restrict__ h) {
  const int b = blockIdx.x >> 2;
  const int hd = blockIdx.x & 3;
  const int t_ = threadIdx.x;
  const int d = t_ & 63;
  const int cb = t_ >> 6;
  float Creg[16];
#pragma unroll
  for (int j = 0; j < 16; ++j) Creg[j] = 0.f;

  __shared__ float sq[64], sk[64], sv[64], so[64], sn[64], sred[64];
  __shared__ float spart[4][64];
  __shared__ float sm, sis, sfs, sden;
  if (t_ < 64) sn[t_] = 0.f;
  if (t_ == 0) sm = 0.f;
  __syncthreads();

  for (int t = 0; t < TT; ++t) {
    if (t_ < 64) {
      size_t base = ((size_t)(b * TT + t)) * HH + hd * DHH + t_;
      sq[t_] = q[base]; sk[t_] = k[base]; sv[t_] = v[base]; so[t_] = o[base];
    }
    if (t_ == 0) {
      size_t gi = (size_t)(b * TT + t) * NHH + hd;
      float it = ip[gi], ft = fp[gi];
      float mn = fmaxf(ft + sm, it);
      sis = __expf(it - mn);
      sfs = __expf(ft + sm - mn);
      sm = mn;
    }
    __syncthreads();
    if (t_ < 64) {                     // n update + denominator partials
      float nn = sfs * sn[t_] + sis * sk[t_];
      sn[t_] = nn;
      sred[t_] = nn * sq[t_];
    }
    __syncthreads();
    if (t_ == 0) {
      float accd = 0.f;
#pragma unroll
      for (int e = 0; e < 64; ++e) accd += sred[e];
      sden = fmaxf(fabsf(accd), 1.0f);
    }
    // C update + num partials (uses sis/sfs/sk/sq/sv, all ready since B1)
    float part = 0.f;
    float vd = sv[d];
    float fs = sfs, is_ = sis;
#pragma unroll
    for (int j = 0; j < 16; ++j) {
      int e = (cb << 4) + j;
      float c = fs * Creg[j] + is_ * vd * sk[e];
      Creg[j] = c;
      part += c * sq[e];
    }
    spart[cb][d] = part;
    __syncthreads();
    if (t_ < 64) {
      float num = spart[0][t_] + spart[1][t_] + spart[2][t_] + spart[3][t_];
      size_t base = ((size_t)(b * TT + t)) * HH + hd * DHH + t_;
      h[base] = so[t_] * num / sden;
    }
    __syncthreads();
  }
}

// ---------------------------------------------------------------------------
// sLSTM sequential scan. One block per (b,head); 256 threads = 4 gate groups
// of 64. Per step: 4 recurrent 64x64 matvecs (R matrices stream from L2,
// fully cache-resident), then per-channel stabilized exp gating.
// ---------------------------------------------------------------------------
__global__ void slstm_scan(const float* __restrict__ zx, const float* __restrict__ ix,
                           const float* __restrict__ fx, const float* __restrict__ ox,
                           const float* __restrict__ Rz, const float* __restrict__ Ri,
                           const float* __restrict__ Rf, const float* __restrict__ Ro,
                           float* __restrict__ hout) {
  const int b = blockIdx.x >> 2;
  const int hd = blockIdx.x & 3;
  const int t_ = threadIdx.x;
  const int g = t_ >> 6;
  const int e = t_ & 63;
  __shared__ float sh[64], sc[64], snv[64], smv[64];
  __shared__ float spre[4][64];
  if (t_ < 64) { sh[t_] = 0.f; sc[t_] = 0.f; snv[t_] = 0.f; smv[t_] = 0.f; }
  __syncthreads();
  const float* Rg = (g == 0 ? Rz : g == 1 ? Ri : g == 2 ? Rf : Ro)
                    + (size_t)hd * DHH * DHH;
  for (int t = 0; t < TT; ++t) {
    float accr = 0.f;                 // (h @ R_g)[e] = sum_d h[d] * R[h][d][e]
#pragma unroll 8
    for (int dd = 0; dd < DHH; ++dd) accr += sh[dd] * Rg[dd * DHH + e];
    spre[g][e] = accr;
    __syncthreads();
    if (t_ < 64) {
      size_t base = ((size_t)(b * TT + t)) * HH + hd * DHH + e;
      float z   = tanhf(zx[base] + spre[0][e]);
      float ipv = ix[base] + spre[1][e];
      float fpv = fx[base] + spre[2][e];
      float ov  = 1.f / (1.f + __expf(-(ox[base] + spre[3][e])));
      float mn  = fmaxf(fpv + smv[e], ipv);
      float is_ = __expf(ipv - mn);
      float fs_ = __expf(fpv + smv[e] - mn);
      float c   = fs_ * sc[e] + is_ * z;
      float nn  = fs_ * snv[e] + is_;
      float hv  = ov * c / nn;
      sc[e] = c; snv[e] = nn; smv[e] = mn; sh[e] = hv;
      hout[base] = hv;
    }
    __syncthreads();
  }
}

// ---------------------------------------------------------------------------
// Final head: out[b] = x[b, T-1, :] @ fc_W + fc_b
// ---------------------------------------------------------------------------
__global__ void fc_kernel(const float* __restrict__ x, const float* __restrict__ fcW,
                          const float* __restrict__ fcb, float* __restrict__ out) {
  int b = threadIdx.x;
  if (b >= BB) return;
  const float* row = x + ((size_t)(b * TT + TT - 1)) * DD;
  float accf = 0.f;
  for (int kk = 0; kk < DD; ++kk) accf += row[kk] * fcW[kk];
  out[b] = accf + fcb[0];
}

// ---------------------------------------------------------------------------
extern "C" void kernel_launch(void* const* d_in, const int* in_sizes, int n_in,
                              void* d_out, int out_size, void* d_ws, size_t ws_size,
                              hipStream_t stream) {
  (void)in_sizes; (void)n_in; (void)out_size; (void)ws_size;
  const float* X      = (const float*)d_in[0];
  const int*   dyad   = (const int*)  d_in[1];
  const float* embed  = (const float*)d_in[2];
  const float* m_ln_g = (const float*)d_in[3];
  const float* m_ln_b = (const float*)d_in[4];
  const float* m_Wq   = (const float*)d_in[5];
  const float* m_Wk   = (const float*)d_in[6];
  const float* m_Wv   = (const float*)d_in[7];
  const float* m_Wi   = (const float*)d_in[8];
  const float* m_Wf   = (const float*)d_in[9];
  const float* m_bi   = (const float*)d_in[10];
  const float* m_bf   = (const float*)d_in[11];
  const float* m_Wo   = (const float*)d_in[12];
  const float* m_Wp   = (const float*)d_in[13];
  const float* s_ln_g = (const float*)d_in[14];
  const float* s_ln_b = (const float*)d_in[15];
  const float* s_Wz   = (const float*)d_in[16];
  const float* s_Wi   = (const float*)d_in[17];
  const float* s_Wf   = (const float*)d_in[18];
  const float* s_Wo   = (const float*)d_in[19];
  const float* s_Rz   = (const float*)d_in[20];
  const float* s_Ri   = (const float*)d_in[21];
  const float* s_Rf   = (const float*)d_in[22];
  const float* s_Ro   = (const float*)d_in[23];
  const float* s_Wp   = (const float*)d_in[24];
  const float* fc_W   = (const float*)d_in[25];
  const float* fc_b   = (const float*)d_in[26];
  float* out = (float*)d_out;

  // Workspace layout (~65 MB): f32 activations then bf16 staging.
  const size_t Mrows = (size_t)BB * TT;                // 8192
  float* w   = (float*)d_ws;
  float* x   = w;                                      // [8192,256]
  float* xn  = x  + Mrows * DD;
  float* qb  = xn + Mrows * DD;
  float* kb  = qb + Mrows * HH;
  float* vb  = kb + Mrows * HH;
  float* ob  = vb + Mrows * HH;
  float* hb  = ob + Mrows * HH;
  float* ipb = hb + Mrows * HH;
  float* fpb = ipb + Mrows * NHH;
  unsigned short* xnb = (unsigned short*)(fpb + Mrows * NHH);
  unsigned short* hbb = xnb + Mrows * DD;
  unsigned int*   wpk = (unsigned int*)(hbb + Mrows * HH); // packed weight slot (32768 dwords)

  dim3 gGrid((unsigned)(Mrows / 16));                  // 512 blocks
  dim3 gBlk(128);                                      // 4 waves, 16x64 strip/wave

  concat_kernel<<<(BB * TT * DD + 255) / 256, 256, 0, stream>>>(X, dyad, embed, x);

  auto pack = [&](const float* Wsrc) {
    pack_w_kernel<<<128, 256, 0, stream>>>(Wsrc, wpk);
  };

  auto run_mlstm = [&](int l) {
    const size_t NW = (size_t)DD * HH;
    ln_kernel<<<(unsigned)Mrows, DD, 0, stream>>>(x, m_ln_g + l * DD, m_ln_b + l * DD, xn, xnb);
    pack(m_Wq + l * NW);
    gemm256_wmma<0><<<gGrid, gBlk, 0, stream>>>(xnb, wpk, qb, nullptr);
    pack(m_Wk + l * NW);
    gemm256_wmma<1><<<gGrid, gBlk, 0, stream>>>(xnb, wpk, kb, nullptr);  // * DH^-0.5
    pack(m_Wv + l * NW);
    gemm256_wmma<0><<<gGrid, gBlk, 0, stream>>>(xnb, wpk, vb, nullptr);
    pack(m_Wo + l * NW);
    gemm256_wmma<2><<<gGrid, gBlk, 0, stream>>>(xnb, wpk, ob, nullptr);  // sigmoid
    ipfp_kernel<<<(unsigned)((Mrows * NHH + 255) / 256), 256, 0, stream>>>(
        xn, m_Wi + (size_t)l * DD * NHH, m_Wf + (size_t)l * DD * NHH,
        m_bi + l * NHH, m_bf + l * NHH, ipb, fpb);
    mlstm_scan<<<BB * NHH, 256, 0, stream>>>(qb, kb, vb, ob, ipb, fpb, hb);
    f2bf_kernel<<<(unsigned)((Mrows * HH + 255) / 256), 256, 0, stream>>>(hb, hbb, (int)(Mrows * HH));
    pack(m_Wp + l * NW);
    gemm256_wmma<3><<<gGrid, gBlk, 0, stream>>>(hbb, wpk, x, x);         // x += h @ Wp
  };

  run_mlstm(0);

  // sLSTM layer
  ln_kernel<<<(unsigned)Mrows, DD, 0, stream>>>(x, s_ln_g, s_ln_b, xn, xnb);
  pack(s_Wz);
  gemm256_wmma<0><<<gGrid, gBlk, 0, stream>>>(xnb, wpk, qb, nullptr);    // zx
  pack(s_Wi);
  gemm256_wmma<0><<<gGrid, gBlk, 0, stream>>>(xnb, wpk, kb, nullptr);    // ix
  pack(s_Wf);
  gemm256_wmma<0><<<gGrid, gBlk, 0, stream>>>(xnb, wpk, vb, nullptr);    // fx
  pack(s_Wo);
  gemm256_wmma<0><<<gGrid, gBlk, 0, stream>>>(xnb, wpk, ob, nullptr);    // ox
  slstm_scan<<<BB * NHH, 256, 0, stream>>>(qb, kb, vb, ob, s_Rz, s_Ri, s_Rf, s_Ro, hb);
  f2bf_kernel<<<(unsigned)((Mrows * HH + 255) / 256), 256, 0, stream>>>(hb, hbb, (int)(Mrows * HH));
  pack(s_Wp);
  gemm256_wmma<3><<<gGrid, gBlk, 0, stream>>>(hbb, wpk, x, x);           // x += h @ Wp

  run_mlstm(1);

  fc_kernel<<<1, 32, 0, stream>>>(x, fc_W, fc_b, out);
}
